// IGCNet_42975442764329
// MI455X (gfx1250) — matrix operand order repaired and checked by reading, not compiled, not run on hardware
//
#include <hip/hip_runtime.h>
#include <hip/hip_bf16.h>
#include <stdint.h>

typedef __attribute__((ext_vector_type(16))) _Float16 v16h;
typedef __attribute__((ext_vector_type(8)))  _Float16 v8h;
typedef __attribute__((ext_vector_type(8)))  float    v8f;

union UF8  { v8f  v; float    f[8]; };
union UH16 { v16h v; v8h      h[2]; };

// ---------------------------------------------------------------------------
// Pack w1_0 (13x64, K zero-padded to 32) and w1_1 (64x64) into WMMA B-operand
// fragment order: frag[f][lane][e] with K = (lane/16)*16 + e, N = ntile*16 + lane%16.
// Per-lane data is contiguous (16 halves = 32B) -> two b128 loads per fragment.
// ---------------------------------------------------------------------------
__global__ void ig_pack_weights(const float* __restrict__ w1_0,
                                const float* __restrict__ w1_1,
                                _Float16* __restrict__ wp0,   // 4 frags * 512
                                _Float16* __restrict__ wp1)   // 8 frags * 512
{
    for (int idx = threadIdx.x; idx < 12 * 512; idx += blockDim.x) {
        int f   = idx >> 9;
        int rem = idx & 511;
        int L   = rem >> 4;
        int e   = rem & 15;
        int lm  = L & 15;
        int lg  = L >> 4;
        if (f < 4) {                       // stage-1 weights (K padded 13->32)
            int K = lg * 16 + e;
            int N = f * 16 + lm;
            float v = (K < 13) ? w1_0[K * 64 + N] : 0.0f;
            wp0[idx] = (_Float16)v;
        } else {                           // stage-2 weights, 2 K-tiles x 4 N-tiles
            int g  = f - 4;
            int kt = g >> 2, nt = g & 3;
            int K  = kt * 32 + lg * 16 + e;
            int N  = nt * 16 + lm;
            wp1[g * 512 + L * 16 + e] = (_Float16)w1_1[K * 64 + N];
        }
    }
}

// mapped(+0.0f) under the order-preserving float->u32 map (msgs are >= 0 post-relu,
// so empty segments resolve to 0.0 exactly like where(isfinite(segment_max), ., 0)).
__global__ void ig_init_aggr(unsigned int* __restrict__ aggr, int n)
{
    int i = blockIdx.x * blockDim.x + threadIdx.x;
    if (i < n) aggr[i] = 0x80000000u;
}

// ---------------------------------------------------------------------------
// Edge kernel: one wave32 per 16-edge tile.
//   h   = relu([x[src] | edge_attr] @ w1_0 + b1_0)   (13->64, 4 WMMAs)
//   msg = relu(h @ w1_1 + b1_1)                      (64->64, 8 WMMAs)
//   aggr[dst] = max(aggr[dst], msg)                  (global_atomic_max_u32)
// ---------------------------------------------------------------------------
__global__ __launch_bounds__(256)
void ig_edge_kernel(const float* __restrict__ xin,
                    const float* __restrict__ eattr,
                    const long long* __restrict__ esrc,
                    const long long* __restrict__ edst,
                    unsigned int* __restrict__ aggr,
                    const _Float16* __restrict__ wp0,
                    const _Float16* __restrict__ wp1,
                    const float* __restrict__ b1_0,
                    const float* __restrict__ b1_1,
                    int nE)
{
    __shared__ __align__(16) _Float16 lds_a[8][16 * 32];
    __shared__ __align__(16) _Float16 lds_h[8][16 * 64];

    const int tid  = threadIdx.x;
    const int wid  = tid >> 5;
    const int lane = tid & 31;
    const int lm   = lane & 15;   // N (or M) index within 16
    const int lg   = lane >> 4;   // lane-group (K half-select)
    const int kb   = lg * 8;

    // Weight fragments resident in VGPRs for the whole wave lifetime (data is
    // ~12KB total, L2-resident; loaded once per wave).
    const v16h* W0 = (const v16h*)wp0;
    const v16h* W1 = (const v16h*)wp1;
    v16h B0[4], B1[8];
#pragma unroll
    for (int nt = 0; nt < 4; ++nt) B0[nt] = W0[nt * 32 + lane];
#pragma unroll
    for (int g = 0; g < 8; ++g)   B1[g]  = W1[g * 32 + lane];

    float bias0[4], bias1[4];
#pragma unroll
    for (int nt = 0; nt < 4; ++nt) {
        bias0[nt] = b1_0[nt * 16 + lm];
        bias1[nt] = b1_1[nt * 16 + lm];
    }

    _Float16* a_lds = lds_a[wid];
    _Float16* h_lds = lds_h[wid];

    const int tiles  = (nE + 15) >> 4;
    const int stride = gridDim.x * 8;
    for (int t = blockIdx.x * 8 + wid; t < tiles; t += stride) {
        const int base = t << 4;

        // ---- stage activation tile into LDS: 16 edges x 32 K (f16, zero pad) ----
        if (lane < 16) {
            int e = base + lane; if (e >= nE) e = nE - 1;
            const long long s = esrc[e];
            const float4 x0 = *(const float4*)(xin + s * 8);
            const float4 x1 = *(const float4*)(xin + s * 8 + 4);
            _Float16* row = a_lds + lane * 32;
            row[0] = (_Float16)x0.x; row[1] = (_Float16)x0.y;
            row[2] = (_Float16)x0.z; row[3] = (_Float16)x0.w;
            row[4] = (_Float16)x1.x; row[5] = (_Float16)x1.y;
            row[6] = (_Float16)x1.z; row[7] = (_Float16)x1.w;
        } else {
            int m = lane - 16;
            int e = base + m; if (e >= nE) e = nE - 1;
            const float* ea = eattr + (long long)e * 5;
            _Float16* row = a_lds + m * 32;
            row[8]  = (_Float16)ea[0]; row[9]  = (_Float16)ea[1];
            row[10] = (_Float16)ea[2]; row[11] = (_Float16)ea[3];
            row[12] = (_Float16)ea[4];
#pragma unroll
            for (int k = 13; k < 32; ++k) row[k] = (_Float16)0.0f;
        }
        asm volatile("s_wait_dscnt 0" ::: "memory");   // RAW through LDS, same wave

        // ---- stage 1: 13->64 ----
        UH16 a;
        a.h[0] = *(const v8h*)(a_lds + lm * 32 + lg * 16);
        a.h[1] = *(const v8h*)(a_lds + lm * 32 + lg * 16 + 8);
        UF8 c[4];
#pragma unroll
        for (int nt = 0; nt < 4; ++nt) {
#pragma unroll
            for (int i = 0; i < 8; ++i) c[nt].f[i] = bias0[nt];
            c[nt].v = __builtin_amdgcn_wmma_f32_16x16x32_f16(
                false, a.v, false, B0[nt], (short)0, c[nt].v, false, false);
        }

        // relu(h) -> LDS as f16 (C-layout rows -> row-major 16x64 tile)
#pragma unroll
        for (int nt = 0; nt < 4; ++nt) {
#pragma unroll
            for (int r = 0; r < 8; ++r) {
                int row = r + (lg << 3);
                float hv = c[nt].f[r];
                hv = hv > 0.0f ? hv : 0.0f;
                h_lds[row * 64 + nt * 16 + lm] = (_Float16)hv;
            }
        }
        asm volatile("s_wait_dscnt 0" ::: "memory");

        // ---- stage 2: 64->64 ----
        UH16 a2[2];
#pragma unroll
        for (int kt = 0; kt < 2; ++kt) {
            a2[kt].h[0] = *(const v8h*)(h_lds + lm * 64 + kt * 32 + lg * 16);
            a2[kt].h[1] = *(const v8h*)(h_lds + lm * 64 + kt * 32 + lg * 16 + 8);
        }
        UF8 c2[4];
#pragma unroll
        for (int nt = 0; nt < 4; ++nt) {
#pragma unroll
            for (int i = 0; i < 8; ++i) c2[nt].f[i] = bias1[nt];
#pragma unroll
            for (int kt = 0; kt < 2; ++kt)
                c2[nt].v = __builtin_amdgcn_wmma_f32_16x16x32_f16(
                    false, a2[kt].v, false, B1[kt * 4 + nt], (short)0, c2[nt].v, false, false);
        }

        // ---- segment-max scatter: order-preserving u32 atomics into L2 ----
        int  dstv[8];
        bool valid[8];
#pragma unroll
        for (int r = 0; r < 8; ++r) {
            int e = base + r + (lg << 3);
            valid[r] = (e < nE);
            dstv[r]  = valid[r] ? (int)edst[e] : 0;
        }
#pragma unroll
        for (int nt = 0; nt < 4; ++nt) {
            int feat = nt * 16 + lm;
#pragma unroll
            for (int r = 0; r < 8; ++r) {
                if (valid[r]) {
                    float m = c2[nt].f[r];
                    m = m > 0.0f ? m : 0.0f;                 // relu; result >= 0
                    unsigned int u = __float_as_uint(m) | 0x80000000u;
                    atomicMax(aggr + (long long)dstv[r] * 64 + feat, u);
                }
            }
        }
    }
}

// ---------------------------------------------------------------------------
// Node kernel: [x | aggr] (72) -> 32 relu -> 5, normalize, rebuild x (8 feats).
// ~0.25 GFLOP per layer -> plain VALU; weights staged in LDS.
// ---------------------------------------------------------------------------
__global__ __launch_bounds__(256)
void ig_node_kernel(const float* __restrict__ xin,
                    const unsigned int* __restrict__ aggr,
                    const float* __restrict__ w2_0, const float* __restrict__ b2_0,
                    const float* __restrict__ w2_1, const float* __restrict__ b2_1,
                    float* __restrict__ xout, int nN)
{
    __shared__ float sw0[72 * 32];
    __shared__ float sw1[32 * 5];
    __shared__ float sb0[32];
    __shared__ float sb1[5];
    for (int i = threadIdx.x; i < 72 * 32; i += 256) sw0[i] = w2_0[i];
    for (int i = threadIdx.x; i < 32 * 5;  i += 256) sw1[i] = w2_1[i];
    if (threadIdx.x < 32) sb0[threadIdx.x] = b2_0[threadIdx.x];
    if (threadIdx.x < 5)  sb1[threadIdx.x] = b2_1[threadIdx.x];
    __syncthreads();

    int n = blockIdx.x * 256 + threadIdx.x;
    if (n >= nN) return;

    float xr[8];
#pragma unroll
    for (int k = 0; k < 8; ++k) xr[k] = xin[(long long)n * 8 + k];

    float h2[32];
#pragma unroll
    for (int j = 0; j < 32; ++j) h2[j] = sb0[j];
#pragma unroll
    for (int k = 0; k < 8; ++k) {
        const float* wr = sw0 + k * 32;
#pragma unroll
        for (int j = 0; j < 32; ++j) h2[j] = fmaf(xr[k], wr[j], h2[j]);
    }
    for (int k = 0; k < 64; ++k) {
        float v = __uint_as_float(aggr[(long long)n * 64 + k] ^ 0x80000000u);
        const float* wr = sw0 + (8 + k) * 32;
#pragma unroll
        for (int j = 0; j < 32; ++j) h2[j] = fmaf(v, wr[j], h2[j]);
    }
#pragma unroll
    for (int j = 0; j < 32; ++j) h2[j] = h2[j] > 0.0f ? h2[j] : 0.0f;

    float comb[5];
#pragma unroll
    for (int m = 0; m < 5; ++m) comb[m] = sb1[m];
    for (int j = 0; j < 32; ++j) {
#pragma unroll
        for (int m = 0; m < 5; ++m) comb[m] = fmaf(h2[j], sw1[j * 5 + m], comb[m]);
    }
    float nor = sqrtf(comb[1] * comb[1] + comb[2] * comb[2] +
                      comb[3] * comb[3] + comb[4] * comb[4]);
    float inv = 1.0f / fmaxf(1.0f, nor);

    float* o = xout + (long long)n * 8;
    o[0] = comb[0];
    o[1] = comb[1] * inv; o[2] = comb[2] * inv;
    o[3] = comb[3] * inv; o[4] = comb[4] * inv;
    o[5] = xr[0]; o[6] = xr[1]; o[7] = xr[2];
}

// ---------------------------------------------------------------------------
extern "C" void kernel_launch(void* const* d_in, const int* in_sizes, int n_in,
                              void* d_out, int out_size, void* d_ws, size_t ws_size,
                              hipStream_t stream)
{
    const float*     x0    = (const float*)d_in[0];
    const float*     eattr = (const float*)d_in[1];
    const long long* eidx  = (const long long*)d_in[2];
    const float*     w1_0  = (const float*)d_in[3];
    const float*     b1_0  = (const float*)d_in[4];
    const float*     w1_1  = (const float*)d_in[5];
    const float*     b1_1  = (const float*)d_in[6];
    const float*     w2_0  = (const float*)d_in[7];
    const float*     b2_0  = (const float*)d_in[8];
    const float*     w2_1  = (const float*)d_in[9];
    const float*     b2_1  = (const float*)d_in[10];

    const int nN = in_sizes[0] / 8;
    const int nE = in_sizes[1] / 5;
    const long long* esrc = eidx;
    const long long* edst = eidx + nE;

    char*  ws  = (char*)d_ws;
    size_t off = 0;
    auto alloc = [&](size_t bytes) {
        size_t o = off;
        off = (off + bytes + 255) & ~(size_t)255;
        return o;
    };
    size_t off_aggr = alloc((size_t)nN * 64 * 4);
    size_t off_xa   = alloc((size_t)nN * 8 * 4);
    size_t off_xb   = alloc((size_t)nN * 8 * 4);
    size_t off_wp0  = alloc(4 * 512 * 2);
    size_t off_wp1  = alloc(8 * 512 * 2);

    unsigned int* aggr = (unsigned int*)(ws + off_aggr);
    float*        xa   = (float*)(ws + off_xa);
    float*        xb   = (float*)(ws + off_xb);
    _Float16*     wp0  = (_Float16*)(ws + off_wp0);
    _Float16*     wp1  = (_Float16*)(ws + off_wp1);

    ig_pack_weights<<<1, 256, 0, stream>>>(w1_0, w1_1, wp0, wp1);

    const int aggrN      = nN * 64;
    const int initBlocks = (aggrN + 255) / 256;
    const int tiles      = (nE + 15) / 16;
    const int edgeBlocks = (tiles + 31) / 32;   // 8 waves/block, ~4 tiles/wave
    const int nodeBlocks = (nN + 255) / 256;

    const float* cur = x0;
    for (int it = 0; it < 8; ++it) {
        float* nxt = (it == 7) ? (float*)d_out : ((it & 1) ? xb : xa);
        ig_init_aggr<<<initBlocks, 256, 0, stream>>>(aggr, aggrN);
        ig_edge_kernel<<<edgeBlocks, 256, 0, stream>>>(cur, eattr, esrc, edst, aggr,
                                                       wp0, wp1, b1_0, b1_1, nE);
        ig_node_kernel<<<nodeBlocks, 256, 0, stream>>>(cur, aggr, w2_0, b2_0,
                                                       w2_1, b2_1, nxt, nN);
        cur = nxt;
    }
}
